// MHSA_89464168776012
// MI455X (gfx1250) — compile-verified, matrix operand order checked by reading, not run on hardware
//
#include <hip/hip_runtime.h>
#include <stdint.h>

typedef __attribute__((ext_vector_type(16))) __bf16 v16bf;
typedef __attribute__((ext_vector_type(8)))  float  v8f;
typedef __attribute__((ext_vector_type(4)))  unsigned int u32x4;
typedef __attribute__((ext_vector_type(8)))  int i32x8;
typedef __attribute__((ext_vector_type(4)))  int i32x4;

union FragU {
    uint4 q[2];   // two 16B halves: elements 0..7 and 8..15
    v16bf v;
};

__device__ __forceinline__ unsigned short f2bf(float f) {
    union { float f; unsigned u; } cv; cv.f = f;
    unsigned u = cv.u;
    unsigned r = (u + 0x7FFFu + ((u >> 16) & 1u)) >> 16;   // RNE
    return (unsigned short)r;
}

// ---------------------------------------------------------------------------
// TDM: async-load a 16-row x 32-col bf16 tile (row stride 512 elements) from
// global into LDS, row-major packed (16*32*2 = 1024 bytes).
// D# group0: count=1 | lds_addr | global_addr | type=2
// D# group1: data_size=2B, tensor_dim0=512, tensor_dim1=32768,
//            tile_dim0=32, tile_dim1=16, tensor_dim0_stride=512
// clang-23 lane: 6-arg builtin (g0, g1, g2, g3, g4, cpol).
// ---------------------------------------------------------------------------
__device__ __forceinline__ void tdm_load_tile_16x32(unsigned lds_byte_addr,
                                                    const void* gaddr) {
    unsigned long long ga = (unsigned long long)(uintptr_t)gaddr;
    u32x4 g0 = { 0x1u,                                   // count=1
                 lds_byte_addr,                          // bits 63:32
                 (unsigned)ga,                           // global_addr lo
                 (unsigned)((ga >> 32) & 0x01FFFFFFu) | (2u << 30) }; // hi|type
    i32x8 g1 = { 0x10000,                 // workgroup_mask=0, data_size=1 (2B)
                 (int)(512u << 16),       // tensor_dim0[15:0] @ bits 63:48
                 (int)0x80000000u,        // tensor_dim1[15:0]=32768 @ bits 95:80
                 (int)(32u << 16),        // tile_dim0=32 @ bits 127:112
                 (int)16,                 // tile_dim1=16 @ bits 143:128
                 (int)512,                // tensor_dim0_stride lo
                 0, 0 };
    i32x4 gz4 = { 0, 0, 0, 0 };
    i32x8 gz8 = { 0, 0, 0, 0, 0, 0, 0, 0 };
    __builtin_amdgcn_tensor_load_to_lds(g0, g1, gz4, gz4, gz8, 0);
}

// ---------------------------------------------------------------------------
// fp32 -> bf16 convert, pair-packed stores
// ---------------------------------------------------------------------------
__global__ void cvt_bf16_kernel(const float* __restrict__ in,
                                unsigned int* __restrict__ out, int npairs) {
    int i = blockIdx.x * blockDim.x + threadIdx.x;
    if (i < npairs) {
        float2 f = ((const float2*)in)[i];
        out[i] = (unsigned)f2bf(f.x) | ((unsigned)f2bf(f.y) << 16);
    }
}

// ---------------------------------------------------------------------------
// Fused QKV GEMM + per-token 8x8 head attention.
// Workgroup: 256 threads (8 waves). Owns 16 tokens, all 1536 QKV outputs.
// A tile (16x32 bf16) staged per k-step via TDM, double-buffered in LDS.
// Wave w covers j in [192w, 192w+192): 12 fragments, B loads pipelined.
// LDS: [0,96KB) fp32 attn scratch; [96KB, 96KB+2KB) A staging buffers.
// ---------------------------------------------------------------------------
__global__ __launch_bounds__(256) void qkv_attn_kernel(
    const unsigned short* __restrict__ xb,
    const unsigned short* __restrict__ wqb,
    const float* __restrict__ bq,
    unsigned short* __restrict__ Y)
{
    extern __shared__ char smem[];
    float* lds = (float*)smem;                                  // 16 x 1536 f32
    unsigned short* astage = (unsigned short*)(smem + 98304);   // 2 x 512 bf16

    const int tid  = threadIdx.x;
    const int wave = tid >> 5;
    const int lane = tid & 31;

    const int tile = blockIdx.x;              // 2048 tiles
    const int b    = tile >> 9;               // batch 0..3
    const int n0   = (tile & 511) << 4;       // token base within batch

    const int ml    = lane & 15;              // M (A) / N (B) within fragment
    const int krow  = (lane >> 4) << 3;       // K-run select per lane half
    const int mhi   = (lane >> 4) << 3;       // C/D: M = r + 8*(lane>=16)
    const int jbase = wave * 192;

    const unsigned short* gA = xb + ((size_t)b * 8192 + n0) * 512;
    const unsigned abase = (unsigned)(uintptr_t)(void*)astage;

    const v8f vzero = {0.f,0.f,0.f,0.f,0.f,0.f,0.f,0.f};
    v8f acc[12];
#pragma unroll
    for (int j = 0; j < 12; ++j) acc[j] = vzero;

    // Prologue: stage k-step 0.
    if (wave == 0) {
        tdm_load_tile_16x32(abase, gA);
        __builtin_amdgcn_s_wait_tensorcnt(0);
    }
    __syncthreads();

    for (int kk = 0; kk < 512; kk += 32) {
        const int cur = (kk >> 5) & 1;
        if (wave == 0 && kk + 32 < 512)
            tdm_load_tile_16x32(abase + (cur ^ 1) * 1024, gA + kk + 32);

        // A fragment from LDS stage (ds_load_b128 x2).
        const unsigned short* abuf = astage + cur * 512;
        FragU a;
        a.q[0] = *(const uint4*)(abuf + ml * 32 + krow);
        a.q[1] = *(const uint4*)(abuf + ml * 32 + krow + 16);

        // Pipelined B loads: fetch j+1 while multiplying j.
        FragU bf[2];
        {
            const unsigned short* brow = wqb + (size_t)(jbase + ml) * 512 + kk;
            bf[0].q[0] = *(const uint4*)(brow + krow);
            bf[0].q[1] = *(const uint4*)(brow + krow + 16);
        }
#pragma unroll
        for (int j = 0; j < 12; ++j) {
            if (j < 11) {
                const unsigned short* brow =
                    wqb + (size_t)(jbase + (j + 1) * 16 + ml) * 512 + kk;
                bf[(j + 1) & 1].q[0] = *(const uint4*)(brow + krow);
                bf[(j + 1) & 1].q[1] = *(const uint4*)(brow + krow + 16);
            }
            acc[j] = __builtin_amdgcn_wmma_f32_16x16x32_bf16(
                false, a.v, false, bf[j & 1].v, (short)0, acc[j], false, false);
        }

        if (wave == 0) __builtin_amdgcn_s_wait_tensorcnt(0);
        __syncthreads();
    }

    // Dump accumulators (+bias) into LDS as (token 16) x (j 1536) fp32.
#pragma unroll
    for (int j = 0; j < 12; ++j) {
        const int col  = jbase + j * 16 + ml;
        const float bias = bq[col];
#pragma unroll
        for (int r = 0; r < 8; ++r) {
            lds[(r + mhi) * 1536 + col] = acc[j][r] + bias;
        }
    }
    __syncthreads();

    // Per-(token, head) attention across heads: 16*8 = 128 pairs.
    if (tid < 128) {
        const int t = tid >> 3;
        const int h = tid & 7;
        const float* qv    = lds + t * 1536 + h * 64;   // q[t,h,:]
        const float* kbase = lds + t * 1536 + 512;      // k[t,g,:]
        const float* vbase = lds + t * 1536 + 1024;     // v[t,g,:]

        float dots[8];
        float mx = -1e30f;
#pragma unroll
        for (int g = 0; g < 8; ++g) {
            const float* kv = kbase + g * 64;
            float s = 0.f;
#pragma unroll
            for (int dd = 0; dd < 64; ++dd) s = fmaf(qv[dd], kv[dd], s);
            s *= 0.125f;                                // d^-0.5, d=64
            s = fminf(50.f, fmaxf(-50.f, s));           // clip
            dots[g] = s;
            mx = fmaxf(mx, s);
        }
        float denom = 0.f;
#pragma unroll
        for (int g = 0; g < 8; ++g) { dots[g] = __expf(dots[g] - mx); denom += dots[g]; }
        const float inv = 1.f / denom;
        float aw[8];
#pragma unroll
        for (int g = 0; g < 8; ++g) aw[g] = dots[g] * inv;

        // Scatter into permuted FC-input layout, bf16, pair-packed stores.
        const int n = n0 + t;
        unsigned short* yrow =
            Y + ((size_t)b * 8192 + h * 1024 + (n >> 3)) * 512 + (n & 7) * 64;
#pragma unroll
        for (int dd = 0; dd < 64; dd += 2) {
            float o0 = 0.f, o1 = 0.f;
#pragma unroll
            for (int g = 0; g < 8; ++g) {
                o0 = fmaf(aw[g], vbase[g * 64 + dd],     o0);
                o1 = fmaf(aw[g], vbase[g * 64 + dd + 1], o1);
            }
            unsigned pack = (unsigned)f2bf(o0) | ((unsigned)f2bf(o1) << 16);
            *(unsigned*)(yrow + dd) = pack;
        }
    }
}

// ---------------------------------------------------------------------------
// FC GEMM: out(32768,512) = Y(32768,512) @ w_fc^T(512,512) + b_fc, fp32 out.
// Workgroup: 8 waves, 16 rows, full 512 cols; A tile staged via TDM.
// ---------------------------------------------------------------------------
__global__ __launch_bounds__(256) void fc_kernel(
    const unsigned short* __restrict__ Y,
    const unsigned short* __restrict__ wfb,
    const float* __restrict__ bfc,
    float* __restrict__ out)
{
    extern __shared__ char smem[];
    unsigned short* astage = (unsigned short*)smem;   // 2 x 512 bf16

    const int tid  = threadIdx.x;
    const int wave = tid >> 5;
    const int lane = tid & 31;

    const size_t m0 = (size_t)blockIdx.x * 16;
    const int ml   = lane & 15;
    const int krow = (lane >> 4) << 3;
    const int mhi  = (lane >> 4) << 3;
    const int nb   = wave * 64;

    const unsigned short* gA = Y + m0 * 512;
    const unsigned abase = (unsigned)(uintptr_t)(void*)astage;

    const v8f vzero = {0.f,0.f,0.f,0.f,0.f,0.f,0.f,0.f};
    v8f acc[4];
#pragma unroll
    for (int j = 0; j < 4; ++j) acc[j] = vzero;

    if (wave == 0) {
        tdm_load_tile_16x32(abase, gA);
        __builtin_amdgcn_s_wait_tensorcnt(0);
    }
    __syncthreads();

    for (int kk = 0; kk < 512; kk += 32) {
        const int cur = (kk >> 5) & 1;
        if (wave == 0 && kk + 32 < 512)
            tdm_load_tile_16x32(abase + (cur ^ 1) * 1024, gA + kk + 32);

        const unsigned short* abuf = astage + cur * 512;
        FragU a;
        a.q[0] = *(const uint4*)(abuf + ml * 32 + krow);
        a.q[1] = *(const uint4*)(abuf + ml * 32 + krow + 16);

        FragU bf[2];
        {
            const unsigned short* brow = wfb + (size_t)(nb + ml) * 512 + kk;
            bf[0].q[0] = *(const uint4*)(brow + krow);
            bf[0].q[1] = *(const uint4*)(brow + krow + 16);
        }
#pragma unroll
        for (int j = 0; j < 4; ++j) {
            if (j < 3) {
                const unsigned short* brow =
                    wfb + (size_t)(nb + (j + 1) * 16 + ml) * 512 + kk;
                bf[(j + 1) & 1].q[0] = *(const uint4*)(brow + krow);
                bf[(j + 1) & 1].q[1] = *(const uint4*)(brow + krow + 16);
            }
            acc[j] = __builtin_amdgcn_wmma_f32_16x16x32_bf16(
                false, a.v, false, bf[j & 1].v, (short)0, acc[j], false, false);
        }

        if (wave == 0) __builtin_amdgcn_s_wait_tensorcnt(0);
        __syncthreads();
    }

#pragma unroll
    for (int j = 0; j < 4; ++j) {
        const int col = nb + j * 16 + ml;
        const float bias = bfc[col];
#pragma unroll
        for (int r = 0; r < 8; ++r) {
            out[(m0 + r + mhi) * 512 + col] = acc[j][r] + bias;
        }
    }
}

// ---------------------------------------------------------------------------
extern "C" void kernel_launch(void* const* d_in, const int* in_sizes, int n_in,
                              void* d_out, int out_size, void* d_ws, size_t ws_size,
                              hipStream_t stream) {
    const float* x     = (const float*)d_in[0];   // (4, 8192, 512)
    const float* w_qkv = (const float*)d_in[1];   // (1536, 512)
    const float* b_qkv = (const float*)d_in[2];   // (1536)
    const float* w_fc  = (const float*)d_in[3];   // (512, 512)
    const float* b_fc  = (const float*)d_in[4];   // (512)
    float* out = (float*)d_out;

    const int NX  = 4 * 8192 * 512;   // 16,777,216
    const int NWQ = 1536 * 512;       //    786,432
    const int NWF = 512 * 512;        //    262,144

    unsigned short* xb  = (unsigned short*)d_ws;
    unsigned short* wqb = xb  + NX;
    unsigned short* wfb = wqb + NWQ;
    unsigned short* Yp  = wfb + NWF;   // total ws ~66 MB

    cvt_bf16_kernel<<<(NX/2  + 255) / 256, 256, 0, stream>>>(x,     (unsigned*)xb,  NX/2);
    cvt_bf16_kernel<<<(NWQ/2 + 255) / 256, 256, 0, stream>>>(w_qkv, (unsigned*)wqb, NWQ/2);
    cvt_bf16_kernel<<<(NWF/2 + 255) / 256, 256, 0, stream>>>(w_fc,  (unsigned*)wfb, NWF/2);

    // 4 batches * 512 token-tiles; 96KB attn scratch + 2KB A staging.
    qkv_attn_kernel<<<2048, 256, 98304 + 2048, stream>>>(xb, wqb, b_qkv, Yp);

    // 32768 rows / 16; 2KB A staging.
    fc_kernel<<<2048, 256, 2048, stream>>>(Yp, wfb, b_fc, out);
}
